// gen_proof_69535520522336
// MI455X (gfx1250) — 8-bit matrix operand order repaired and checked by reading, compiled, not run on hardware
//
#include <hip/hip_runtime.h>
#include <stdint.h>

#define P32 2013265921u
static constexpr int LOGN = 20;
static constexpr int NN   = 1 << LOGN;

__device__ __forceinline__ uint32_t addm(uint32_t a, uint32_t b) {
    uint32_t s = a + b; return (s >= P32) ? s - P32 : s;
}
__device__ __forceinline__ uint32_t subm(uint32_t a, uint32_t b) {
    return (a >= b) ? a - b : a + P32 - b;
}
__device__ __forceinline__ uint32_t mulm(uint32_t a, uint32_t b) {
    return (uint32_t)(((unsigned long long)a * b) % P32);
}
__device__ uint32_t powm(uint32_t b, uint32_t e) {
    uint32_t r = 1u;
    while (e) { if (e & 1u) r = mulm(r, b); b = mulm(b, b); e >>= 1u; }
    return r;
}
__device__ __forceinline__ uint32_t invm(uint32_t x) { return powm(x, P32 - 2u); }
__device__ __forceinline__ uint32_t rootW() { return powm(31u, 1920u); }

__global__ void k_twiddles(uint32_t* __restrict__ tw) {
    unsigned i = blockIdx.x * blockDim.x + threadIdx.x;
    if (i == 0u) { tw[0] = 1u; return; }
    unsigned half = 1u << (31 - __clz((int)i));
    unsigned k    = i - half;
    unsigned m    = half << 1;
    unsigned e    = (unsigned)(((unsigned long long)(NN / m) * k) % (unsigned)NN);
    e = ((unsigned)NN - e) % (unsigned)NN;
    tw[i] = powm(rootW(), e);
}

__global__ void k_bitrev(uint32_t* __restrict__ dst, const uint32_t* __restrict__ src,
                         unsigned stride, unsigned offset) {
    unsigned i = blockIdx.x * blockDim.x + threadIdx.x;
    unsigned r = __brev(i) >> (32 - LOGN);
    dst[i] = src[(size_t)offset + (size_t)stride * r];
}

__global__ void __launch_bounds__(512) k_ntt_lds(uint32_t* __restrict__ a,
                                                 const uint32_t* __restrict__ tw) {
    __shared__ uint32_t sh[1024];
    unsigned tid  = threadIdx.x;
    size_t   base = (size_t)blockIdx.x * 1024;
    sh[tid]       = a[base + tid];
    sh[tid + 512] = a[base + tid + 512];
    __syncthreads();
#pragma unroll
    for (unsigned s = 0; s < 10; ++s) {
        unsigned half = 1u << s;
        unsigned grp  = tid >> s;
        unsigned k    = tid & (half - 1u);
        unsigned pos  = grp * (half << 1) + k;
        uint32_t u  = sh[pos];
        uint32_t tv = mulm(sh[pos + half], tw[half + k]);
        sh[pos]        = addm(u, tv);
        sh[pos + half] = subm(u, tv);
        __syncthreads();
    }
    a[base + tid]       = sh[tid];
    a[base + tid + 512] = sh[tid + 512];
}

__global__ void k_ntt_stage(uint32_t* __restrict__ a, const uint32_t* __restrict__ tw,
                            unsigned s) {
    unsigned tid  = blockIdx.x * blockDim.x + threadIdx.x;
    unsigned half = 1u << s;
    unsigned grp  = tid >> s;
    unsigned k    = tid & (half - 1u);
    size_t   pos  = (size_t)grp * (half << 1) + k;
    uint32_t u  = a[pos];
    uint32_t tv = mulm(a[pos + half], tw[half + k]);
    a[pos]        = addm(u, tv);
    a[pos + half] = subm(u, tv);
}

__global__ void k_finalize(uint32_t* __restrict__ a, float* __restrict__ out) {
    unsigned i = blockIdx.x * blockDim.x + threadIdx.x;
    uint32_t y = mulm(a[i], invm((uint32_t)NN));
    a[i]   = y;
    out[i] = (float)y;
}

__global__ void k_combine_t(uint32_t* __restrict__ t,
                            const uint32_t* __restrict__ t1, const uint32_t* __restrict__ t2,
                            const uint32_t* __restrict__ t3, const uint32_t* __restrict__ t4,
                            const int* __restrict__ zp) {
    unsigned i = blockIdx.x * blockDim.x + threadIdx.x;
    uint32_t z  = (uint32_t)zp[0] % P32;
    uint32_t z2 = mulm(z, z), z3 = mulm(z2, z);
    t[i] = addm(addm(t1[i], mulm(z, t2[i])), addm(mulm(z2, t3[i]), mulm(z3, t4[i])));
}

__global__ void k_make_f(uint32_t* __restrict__ f,
                         const uint32_t* __restrict__ wl, const uint32_t* __restrict__ wr,
                         const uint32_t* __restrict__ wo, const uint32_t* __restrict__ w4,
                         const uint32_t* __restrict__ ql, const uint32_t* __restrict__ t,
                         const int* __restrict__ zp) {
    unsigned i = blockIdx.x * blockDim.x + threadIdx.x;
    uint32_t z  = (uint32_t)zp[0] % P32;
    uint32_t z2 = mulm(z, z), z3 = mulm(z2, z);
    uint32_t q = addm(addm(wl[i], mulm(z, wr[i])), addm(mulm(z2, wo[i]), mulm(z3, w4[i])));
    f[i] = (ql[i] == 1u) ? q : t[i];
}

__global__ void k_fill_s(uint32_t* __restrict__ s, const uint32_t* __restrict__ f,
                         const uint32_t* __restrict__ t) {
    unsigned i = blockIdx.x * blockDim.x + threadIdx.x;
    s[i]      = f[i];
    s[NN + i] = t[i];
}

__device__ __forceinline__ void cswap(uint32_t& x, uint32_t& y, bool up) {
    uint32_t lo = (x < y) ? x : y;
    uint32_t hi = (x < y) ? y : x;
    x = up ? lo : hi;
    y = up ? hi : lo;
}

__global__ void __launch_bounds__(1024) k_bitonic_local_start(uint32_t* __restrict__ s) {
    __shared__ uint32_t sh[2048];
    unsigned tid  = threadIdx.x;
    size_t   base = (size_t)blockIdx.x * 2048;
    sh[tid]        = s[base + tid];
    sh[tid + 1024] = s[base + tid + 1024];
    __syncthreads();
    for (unsigned k = 2; k <= 2048; k <<= 1) {
        for (unsigned j = k >> 1; j > 0; j >>= 1) {
            unsigned idx = 2u * tid - (tid & (j - 1u));
            bool up = (((base + idx) & (size_t)k) == 0);
            uint32_t x = sh[idx], y = sh[idx + j];
            cswap(x, y, up);
            sh[idx] = x; sh[idx + j] = y;
            __syncthreads();
        }
    }
    s[base + tid]        = sh[tid];
    s[base + tid + 1024] = sh[tid + 1024];
}

__global__ void __launch_bounds__(1024) k_bitonic_local_merge(uint32_t* __restrict__ s,
                                                              unsigned k) {
    __shared__ uint32_t sh[2048];
    unsigned tid  = threadIdx.x;
    size_t   base = (size_t)blockIdx.x * 2048;
    sh[tid]        = s[base + tid];
    sh[tid + 1024] = s[base + tid + 1024];
    __syncthreads();
    for (unsigned j = 1024; j > 0; j >>= 1) {
        unsigned idx = 2u * tid - (tid & (j - 1u));
        bool up = (((base + idx) & (size_t)k) == 0);
        uint32_t x = sh[idx], y = sh[idx + j];
        cswap(x, y, up);
        sh[idx] = x; sh[idx + j] = y;
        __syncthreads();
    }
    s[base + tid]        = sh[tid];
    s[base + tid + 1024] = sh[tid + 1024];
}

__global__ void k_bitonic_global(uint32_t* __restrict__ s, unsigned k, unsigned j) {
    unsigned tid = blockIdx.x * blockDim.x + threadIdx.x;
    unsigned idx = 2u * tid - (tid & (j - 1u));
    bool up = ((idx & k) == 0);
    uint32_t x = s[idx], y = s[idx + j];
    cswap(x, y, up);
    s[idx] = x; s[idx + j] = y;
}

__global__ void k_perm_r(uint32_t* __restrict__ r,
                         const uint32_t* __restrict__ wl, const uint32_t* __restrict__ wr,
                         const uint32_t* __restrict__ wo, const uint32_t* __restrict__ w4,
                         const uint32_t* __restrict__ sl, const uint32_t* __restrict__ sr,
                         const uint32_t* __restrict__ so, const uint32_t* __restrict__ s4,
                         const int* __restrict__ bp, const int* __restrict__ gp) {
    unsigned i = blockIdx.x * blockDim.x + threadIdx.x;
    uint32_t beta  = (uint32_t)bp[0] % P32;
    uint32_t gamma = (uint32_t)gp[0] % P32;
    uint32_t om = powm(rootW(), i);
    uint32_t id0 = om, id1 = mulm(7u, om), id2 = mulm(13u, om), id3 = mulm(17u, om);
    uint32_t num = 1u, den = 1u;
    num = mulm(num, addm(addm(wl[i], mulm(beta, id0)), gamma));
    den = mulm(den, addm(addm(wl[i], mulm(beta, sl[i])), gamma));
    num = mulm(num, addm(addm(wr[i], mulm(beta, id1)), gamma));
    den = mulm(den, addm(addm(wr[i], mulm(beta, sr[i])), gamma));
    num = mulm(num, addm(addm(wo[i], mulm(beta, id2)), gamma));
    den = mulm(den, addm(addm(wo[i], mulm(beta, so[i])), gamma));
    num = mulm(num, addm(addm(w4[i], mulm(beta, id3)), gamma));
    den = mulm(den, addm(addm(w4[i], mulm(beta, s4[i])), gamma));
    r[i] = mulm(num, invm(den));
}

__global__ void k_z2_r(uint32_t* __restrict__ r,
                       const uint32_t* __restrict__ f, const uint32_t* __restrict__ t,
                       const uint32_t* __restrict__ s,
                       const int* __restrict__ dp, const int* __restrict__ ep) {
    unsigned i  = blockIdx.x * blockDim.x + threadIdx.x;
    unsigned in = (i + 1u) & (unsigned)(NN - 1);
    uint32_t delta = (uint32_t)dp[0] % P32;
    uint32_t eps   = (uint32_t)ep[0] % P32;
    uint32_t opd  = addm(1u, delta);
    uint32_t ep1d = mulm(eps, opd);
    uint32_t h1  = s[2 * (size_t)i],  h2  = s[2 * (size_t)i + 1];
    uint32_t h1n = s[2 * (size_t)in], h2n = s[2 * (size_t)in + 1];
    uint32_t num = mulm(opd, addm(eps, f[i]));
    num = mulm(num, addm(ep1d, addm(t[i], mulm(delta, t[in]))));
    uint32_t a = addm(ep1d, addm(h1, mulm(delta, h1n)));
    uint32_t b = addm(ep1d, addm(h2, mulm(delta, h2n)));
    r[i] = mulm(num, invm(mulm(a, b)));
}

__global__ void __launch_bounds__(1024) k_scan1(const uint32_t* __restrict__ r,
                                                uint32_t* __restrict__ tmp,
                                                uint32_t* __restrict__ bsum) {
    __shared__ uint32_t sh[1024];
    unsigned tid = threadIdx.x;
    unsigned gi  = blockIdx.x * 1024u + tid;
    sh[tid] = r[gi];
    __syncthreads();
    for (unsigned off = 1; off < 1024; off <<= 1) {
        uint32_t v = (tid >= off) ? sh[tid - off] : 1u;
        __syncthreads();
        if (tid >= off) sh[tid] = mulm(sh[tid], v);
        __syncthreads();
    }
    tmp[gi] = sh[tid];
    if (tid == 1023u) bsum[blockIdx.x] = sh[1023];
}

__global__ void __launch_bounds__(1024) k_scan2(const uint32_t* __restrict__ bsum,
                                                uint32_t* __restrict__ bexcl) {
    __shared__ uint32_t sh[1024];
    unsigned tid = threadIdx.x;
    sh[tid] = bsum[tid];
    __syncthreads();
    for (unsigned off = 1; off < 1024; off <<= 1) {
        uint32_t v = (tid >= off) ? sh[tid - off] : 1u;
        __syncthreads();
        if (tid >= off) sh[tid] = mulm(sh[tid], v);
        __syncthreads();
    }
    bexcl[tid] = (tid == 0u) ? 1u : sh[tid - 1];
}

__global__ void __launch_bounds__(1024) k_scan3(const uint32_t* __restrict__ tmp,
                                                const uint32_t* __restrict__ bexcl,
                                                uint32_t* __restrict__ z) {
    unsigned tid = threadIdx.x;
    unsigned gi  = blockIdx.x * 1024u + tid;
    uint32_t local = (tid == 0u) ? 1u : tmp[gi - 1];
    z[gi] = mulm(bexcl[blockIdx.x], local);
}

typedef __attribute__((ext_vector_type(8))) int v8i_t;

static constexpr unsigned COMMIT_BLOCKS = 64;
static constexpr unsigned COMMIT_TPB    = 256;
static constexpr unsigned COMMIT_WAVES  = COMMIT_BLOCKS * (COMMIT_TPB / 32);
static constexpr unsigned COMMIT_GROUPS = (unsigned)NN >> 8;
static constexpr unsigned COMMIT_ITERS  = COMMIT_GROUPS / COMMIT_WAVES;

__global__ void __launch_bounds__(COMMIT_TPB) k_commit_wmma(
        const uint32_t* __restrict__ poly, const uint32_t* __restrict__ ck,
        unsigned long long* __restrict__ accum) {
    __shared__ uint32_t shP[COMMIT_TPB / 32][256];
    __shared__ uint32_t shC[COMMIT_TPB / 32][256];
    const unsigned lane  = threadIdx.x & 31u;
    const unsigned wv    = threadIdx.x >> 5;
    const unsigned waveG = blockIdx.x * (COMMIT_TPB / 32) + wv;
    const unsigned col   = lane & 15u;
    const unsigned hi    = lane >> 4;
    const unsigned chunkR = col >> 2, limbR = col & 3u;
    const uint32_t ldsP = (uint32_t)(uintptr_t)&shP[wv][lane];
    const uint32_t ldsC = (uint32_t)(uintptr_t)&shC[wv][lane];

    v8i_t c = {0, 0, 0, 0, 0, 0, 0, 0};
#pragma unroll
    for (unsigned it = 0; it < COMMIT_ITERS; ++it) {
        const unsigned grp  = waveG + it * COMMIT_WAVES;
        const size_t   base = (size_t)grp << 8;
        const uint32_t* gp_ = poly + base + lane;
        const uint32_t* gc_ = ck   + base + lane;
        asm volatile("s_wait_dscnt 0x0" ::: "memory");
#pragma unroll
        for (int q = 0; q < 8; ++q) {
            asm volatile("global_load_async_to_lds_b32 %0, %1, off offset:%2"
                         :: "v"(ldsP), "v"(gp_), "i"(128 * q) : "memory");
            asm volatile("global_load_async_to_lds_b32 %0, %1, off offset:%2"
                         :: "v"(ldsC), "v"(gc_), "i"(128 * q) : "memory");
        }
        __builtin_prefetch(&poly[base + (size_t)COMMIT_WAVES * 256], 0, 0);
        asm volatile("s_wait_asynccnt 0x0" ::: "memory");

        v8i_t a, b;
#pragma unroll
        for (int v = 0; v < 8; ++v) {
            uint32_t wa = 0u, wb = 0u;
#pragma unroll
            for (int by = 0; by < 4; ++by) {
                unsigned ka = ((unsigned)(v >> 1) << 4) + ((unsigned)(v & 1) << 2)
                            + (unsigned)by + (hi ? 8u : 0u);
                unsigned kb = ((unsigned)(v >> 1) << 4) + ((unsigned)(v & 1) << 2)
                            + (unsigned)by + (hi ? 8u : 0u);
                uint32_t ba = (shP[wv][(chunkR << 6) + ka] >> (8u * limbR)) & 0xFFu;
                uint32_t bb = (shC[wv][(chunkR << 6) + kb] >> (8u * limbR)) & 0xFFu;
                wa |= ba << (8 * by);
                wb |= bb << (8 * by);
            }
            a[v] = (int)wa;
            b[v] = (int)wb;
        }
        c = __builtin_amdgcn_wmma_i32_16x16x64_iu8(false, a, false, b, c, false, false);
    }
    const unsigned needHi = chunkR >> 1;
    const uint32_t mask   = (hi == needHi) ? 0xFFFFFFFFu : 0u;
    const unsigned r0     = (chunkR & 1u) << 2;
    uint32_t partial = 0u;
#pragma unroll
    for (unsigned i = 0; i < 4; ++i) {
        uint32_t cv = (uint32_t)(((unsigned)c[r0 + i] & mask) % P32);
        partial = addm(partial, mulm(cv, powm(2u, 8u * (i + limbR))));
    }
    for (int off = 16; off > 0; off >>= 1)
        partial = addm(partial, (uint32_t)__shfl_xor((int)partial, off, 32));
    if (lane == 0u)
        atomicAdd(accum, (unsigned long long)partial);
}

__global__ void k_zero_accum(unsigned long long* __restrict__ acc) {
    if (threadIdx.x < 10u) acc[threadIdx.x] = 0ull;
}
__global__ void k_write_commits(const unsigned long long* __restrict__ acc,
                                float* __restrict__ out) {
    if (threadIdx.x < 10u) out[threadIdx.x] = (float)(acc[threadIdx.x] % (unsigned long long)P32);
}

extern "C" void kernel_launch(void* const* d_in, const int* in_sizes, int n_in,
                              void* d_out, int out_size, void* d_ws, size_t ws_size,
                              hipStream_t stream) {
    (void)in_sizes; (void)n_in; (void)out_size; (void)ws_size;
    const uint32_t* wl = (const uint32_t*)d_in[0];
    const uint32_t* wr = (const uint32_t*)d_in[1];
    const uint32_t* wo = (const uint32_t*)d_in[2];
    const uint32_t* w4 = (const uint32_t*)d_in[3];
    const uint32_t* ql = (const uint32_t*)d_in[4];
    const uint32_t* t1 = (const uint32_t*)d_in[5];
    const uint32_t* t2 = (const uint32_t*)d_in[6];
    const uint32_t* t3 = (const uint32_t*)d_in[7];
    const uint32_t* t4 = (const uint32_t*)d_in[8];
    const uint32_t* sl = (const uint32_t*)d_in[9];
    const uint32_t* sr = (const uint32_t*)d_in[10];
    const uint32_t* so = (const uint32_t*)d_in[11];
    const uint32_t* s4 = (const uint32_t*)d_in[12];
    const uint32_t* ckp = (const uint32_t*)d_in[13];
    const int* zp = (const int*)d_in[14];
    const int* bp = (const int*)d_in[15];
    const int* gp = (const int*)d_in[16];
    const int* dp = (const int*)d_in[17];
    const int* ep = (const int*)d_in[18];
    float* out = (float*)d_out;

    uint32_t* ws   = (uint32_t*)d_ws;
    uint32_t* WNTT = ws;
    uint32_t* T    = ws + (size_t)NN;
    uint32_t* F    = ws + 2*(size_t)NN;
    uint32_t* S    = ws + 3*(size_t)NN;
    uint32_t* R    = ws + 5*(size_t)NN;
    uint32_t* SC   = ws + 6*(size_t)NN;
    uint32_t* Z    = ws + 7*(size_t)NN;
    uint32_t* Z2   = ws + 8*(size_t)NN;
    uint32_t* TW   = ws + 9*(size_t)NN;
    uint32_t* BS   = ws + 10*(size_t)NN;
    uint32_t* BE   = BS + 1024;
    unsigned long long* ACC = (unsigned long long*)(ws + 10*(size_t)NN + 4096);

    const int TPB = 256;
    const int GN  = NN / TPB;

    k_zero_accum<<<1, 32, 0, stream>>>(ACC);
    k_twiddles<<<GN, TPB, 0, stream>>>(TW);
    k_combine_t<<<GN, TPB, 0, stream>>>(T, t1, t2, t3, t4, zp);
    k_make_f<<<GN, TPB, 0, stream>>>(F, wl, wr, wo, w4, ql, T, zp);
    k_fill_s<<<GN, TPB, 0, stream>>>(S, F, T);

    const unsigned M = 2u * (unsigned)NN;
    k_bitonic_local_start<<<M / 2048, 1024, 0, stream>>>(S);
    for (unsigned k = 4096; k <= M; k <<= 1) {
        for (unsigned j = k >> 1; j >= 2048; j >>= 1)
            k_bitonic_global<<<(M / 2) / TPB, TPB, 0, stream>>>(S, k, j);
        k_bitonic_local_merge<<<M / 2048, 1024, 0, stream>>>(S, k);
    }

    k_perm_r<<<GN, TPB, 0, stream>>>(R, wl, wr, wo, w4, sl, sr, so, s4, bp, gp);
    k_scan1<<<NN / 1024, 1024, 0, stream>>>(R, SC, BS);
    k_scan2<<<1, 1024, 0, stream>>>(BS, BE);
    k_scan3<<<NN / 1024, 1024, 0, stream>>>(SC, BE, Z);

    k_z2_r<<<GN, TPB, 0, stream>>>(R, F, T, S, dp, ep);
    k_scan1<<<NN / 1024, 1024, 0, stream>>>(R, SC, BS);
    k_scan2<<<1, 1024, 0, stream>>>(BS, BE);
    k_scan3<<<NN / 1024, 1024, 0, stream>>>(SC, BE, Z2);

    struct Src { const uint32_t* p; unsigned stride, off; };
    const Src srcs[10] = {
        {wl, 1, 0}, {wr, 1, 0}, {wo, 1, 0}, {w4, 1, 0},
        {F, 1, 0},  {T, 1, 0},  {S, 2, 0},  {S, 2, 1},
        {Z, 1, 0},  {Z2, 1, 0}
    };
    for (int p = 0; p < 10; ++p) {
        k_bitrev<<<GN, TPB, 0, stream>>>(WNTT, srcs[p].p, srcs[p].stride, srcs[p].off);
        k_ntt_lds<<<NN / 1024, 512, 0, stream>>>(WNTT, TW);
        for (unsigned s = 10; s < 20; ++s)
            k_ntt_stage<<<(NN / 2) / TPB, TPB, 0, stream>>>(WNTT, TW, s);
        k_finalize<<<GN, TPB, 0, stream>>>(WNTT, out + (size_t)p * NN);
        k_commit_wmma<<<COMMIT_BLOCKS, COMMIT_TPB, 0, stream>>>(WNTT, ckp, ACC + p);
    }
    k_write_commits<<<1, 32, 0, stream>>>(ACC, out + (size_t)10 * NN);
}
